// MultiHeadAttention_13374528160453
// MI455X (gfx1250) — matrix operand order repaired and checked by reading, not compiled, not run on hardware
//
#include <hip/hip_runtime.h>

// ---------------- problem constants ----------------
#define NE   384
#define NH   6
#define NKV  3
#define HD   64
#define BB   8
#define TT   2048
#define MM   (BB * TT)          // 16384 rows

typedef __attribute__((ext_vector_type(16))) __bf16 bf16x16;
typedef __attribute__((ext_vector_type(8)))  __bf16 bf16x8;
typedef __attribute__((ext_vector_type(4)))  __bf16 bf16x4;
typedef __attribute__((ext_vector_type(8)))  float  f32x8;
typedef __attribute__((ext_vector_type(4)))  float  f32x4;

union AFrag { bf16x16 v; bf16x8 h[2]; };

__device__ __forceinline__ f32x8 wmma_bf16(bf16x16 a, bf16x16 b, f32x8 c) {
  // (neg_a, A, neg_b, B, c_mod, C, reuse_a, reuse_b)
  return __builtin_amdgcn_wmma_f32_16x16x32_bf16(false, a, false, b, (short)0, c, false, false);
}

__device__ __forceinline__ float fast_exp2(float x) {
#if __has_builtin(__builtin_amdgcn_exp2f)
  return __builtin_amdgcn_exp2f(x);     // raw v_exp_f32
#else
  return __expf(x * 0.6931471805599453f);
#endif
}

// ---------------- fp32 -> bf16 convert (x and weights) ----------------
__global__ void cvt_f32_bf16(const float* __restrict__ in, __bf16* __restrict__ out, int n4) {
  int i = blockIdx.x * blockDim.x + threadIdx.x;
  if (i < n4) {
    f32x4 v = ((const f32x4*)in)[i];
    ((bf16x4*)out)[i] = __builtin_convertvector(v, bf16x4);
  }
}

// ---------------- generic WMMA GEMM: y[m,n] = sum_k A[m,k] * W[n,k] + bias[n] ----
// MODE 0: out = float [M,N] row-major              (final projection)
// MODE 1: out = bf16 scattered [b, n/64, t, n%64]  (Q, K head-major)
// MODE 2: out = bf16 scattered [b, n/64, n%64, t]  (V transposed, d-major)
template <int MODE>
__global__ void gemm_xWT(const __bf16* __restrict__ A, const __bf16* __restrict__ W,
                         const float* __restrict__ bias, void* __restrict__ outp,
                         int M, int N, int K, int heads) {
  const int lane = threadIdx.x & 31;
  const int r    = lane & 15;
  const int hi   = lane >> 4;
  const int wid  = (blockIdx.x * blockDim.x + threadIdx.x) >> 5;
  const int ntn  = N >> 6;                       // 64-wide column tiles
  const int nTiles = (M >> 4) * ntn;
  if (wid >= nTiles) return;                     // wave-uniform: EXEC stays full
  const int mb = (wid / ntn) << 4;
  const int nb = (wid % ntn) << 6;

  f32x8 acc[4] = {};
  const __bf16* arow = A + (size_t)(mb + r) * K;
  for (int k0 = 0; k0 < K; k0 += 32) {
    AFrag a;                                     // A 16x32: two 16B contiguous chunks
    a.h[0] = *(const bf16x8*)(arow + k0 + 16 * hi);
    a.h[1] = *(const bf16x8*)(arow + k0 + 16 * hi + 8);
#pragma unroll
    for (int s = 0; s < 4; ++s) {                // B 32x16: one 32B contiguous chunk
      bf16x16 b = *(const bf16x16*)(W + (size_t)(nb + 16 * s + r) * K + k0 + 16 * hi);
      acc[s] = wmma_bf16(a.v, b, acc[s]);
    }
  }

#pragma unroll
  for (int s = 0; s < 4; ++s) {
    const int n = nb + 16 * s + r;
    const float bv = bias[n];
#pragma unroll
    for (int j = 0; j < 8; ++j) {
      const int m = mb + j + 8 * hi;
      const float v = acc[s][j] + bv;
      if (MODE == 0) {
        ((float*)outp)[(size_t)m * N + n] = v;
      } else {
        const int bi = m >> 11;                  // m / TT
        const int t  = m & (TT - 1);
        const int h  = n >> 6;
        const int d  = n & 63;
        if (MODE == 1)
          ((__bf16*)outp)[(((size_t)bi * heads + h) * TT + t) * HD + d] = (__bf16)v;
        else
          ((__bf16*)outp)[(((size_t)bi * heads + h) * HD + d) * TT + t] = (__bf16)v;
      }
    }
  }
}

// ---------------- flash attention: one wave per (b, h, 16-query tile) ----------
// Softmax in row-owner layout via one LDS transpose (padded, conflict-free).
// Chunk loop split: full-visibility chunks carry zero mask code; exactly one
// trailing chunk per wave runs the causal-edge path.
#define SPAD 36   // f32 S-tile row stride (36*4B -> conflict-free b128 row reads)
#define PPAD 40   // bf16 P-tile row stride (40*2B -> conflict-free, 16B aligned)

__global__ void flash_attn(const __bf16* __restrict__ qb_, const __bf16* __restrict__ kb_,
                           const __bf16* __restrict__ vtb_, __bf16* __restrict__ attnb,
                           float sl2 /* scale * log2(e) */) {
  __shared__ __align__(16) float  sls[8][16][SPAD]; // raw/masked S tile (f32)
  __shared__ __align__(16) __bf16 pls[8][16][PPAD]; // P tile (bf16, A-frag source)
  __shared__ __align__(16) float  cls[8][16];       // per-row broadcast (corr / inv-l)

  const int lane = threadIdx.x & 31;
  const int r    = lane & 15;
  const int hi   = lane >> 4;
  const int wv   = threadIdx.x >> 5;
  const int wid  = (blockIdx.x * blockDim.x + threadIdx.x) >> 5;
  const int qt   = wid & (TT / 16 - 1);             // 128 query tiles
  const int h    = (wid >> 7) % NH;
  const int b    = wid / ((TT / 16) * NH);
  const int kv   = h >> 1;                          // repeat_kv: head h -> kv h/2
  const int qbrow = qt << 4;

  const __bf16* qptr = qb_  + (((size_t)b * NH  + h ) * TT + qbrow) * HD;
  const __bf16* kptr = kb_  + (((size_t)b * NKV + kv) * TT) * HD;
  const __bf16* vptr = vtb_ + (((size_t)b * NKV + kv) * HD) * TT;

  AFrag qa[2];                                      // Q 16x64 as two K=32 A-frags
#pragma unroll
  for (int s = 0; s < 2; ++s) {
    qa[s].h[0] = *(const bf16x8*)(qptr + r * HD + 32 * s + 16 * hi);
    qa[s].h[1] = *(const bf16x8*)(qptr + r * HD + 32 * s + 16 * hi + 8);
  }

  f32x8 o[4] = {};
  float rm = -3.0e38f;                              // running max (raw scores), row r
  float rl = 0.0f;                                  // running sum, row r

  auto do_chunk = [&](int c, bool edge) __attribute__((always_inline)) {
    f32x8 s0 = {}, s1 = {};
#pragma unroll
    for (int s = 0; s < 2; ++s) {                   // S = Q K^T over d (two K=32 steps)
      bf16x16 b0 = *(const bf16x16*)(kptr + (size_t)(c + r)      * HD + 32 * s + 16 * hi);
      bf16x16 b1 = *(const bf16x16*)(kptr + (size_t)(c + 16 + r) * HD + 32 * s + 16 * hi);
      s0 = wmma_bf16(qa[s].v, b0, s0);
      s1 = wmma_bf16(qa[s].v, b1, s1);
    }
    // causal mask (edge chunk only; dead-coded in the hot loop), then transpose
#pragma unroll
    for (int j = 0; j < 8; ++j) {
      float a0 = s0[j], a1 = s1[j];
      if (edge) {
        const int qrow = qbrow + j + 8 * hi;
        a0 = (c + r      > qrow) ? -3.0e38f : a0;
        a1 = (c + 16 + r > qrow) ? -3.0e38f : a1;
      }
      sls[wv][j + 8 * hi][r]      = a0;
      sls[wv][j + 8 * hi][16 + r] = a1;
    }
    // row-owner softmax: lane handles (row=r, half=hi) -> 16 contiguous f32
    f32x4 rv[4];
#pragma unroll
    for (int u = 0; u < 4; ++u)
      rv[u] = ((const f32x4*)&sls[wv][r][16 * hi])[u];
    float lm = -3.0e38f;
#pragma unroll
    for (int u = 0; u < 4; ++u)
#pragma unroll
      for (int e = 0; e < 4; ++e) lm = fmaxf(lm, rv[u][e]);
    lm = fmaxf(lm, __shfl_xor(lm, 16));             // merge halves (1 ds op)
    const float mnew = fmaxf(rm, lm);
    const float cor  = fast_exp2((rm - mnew) * sl2);
    float ls = 0.0f;
    f32x8 elo, ehi;
#pragma unroll
    for (int u = 0; u < 4; ++u)
#pragma unroll
      for (int e = 0; e < 4; ++e) {
        const float ev = fast_exp2((rv[u][e] - mnew) * sl2);
        ls += ev;
        if (u < 2) elo[u * 4 + e] = ev; else ehi[(u - 2) * 4 + e] = ev;
      }
    ls += __shfl_xor(ls, 16);                       // merge halves (1 ds op)
    rl = rl * cor + ls;
    rm = mnew;
    cls[wv][r] = cor;                               // both halves write same value
    // P (bf16) straight into A-frag source layout: 2 contiguous 16B stores
    ((bf16x8*)&pls[wv][r][0])[2 * hi]     = __builtin_convertvector(elo, bf16x8);
    ((bf16x8*)&pls[wv][r][0])[2 * hi + 1] = __builtin_convertvector(ehi, bf16x8);
    // rescale O accumulators with per-row corr, broadcast via LDS (C layout rows)
    const f32x4 ca = ((const f32x4*)&cls[wv][8 * hi])[0];
    const f32x4 cb = ((const f32x4*)&cls[wv][8 * hi])[1];
#pragma unroll
    for (int t = 0; t < 4; ++t)
#pragma unroll
      for (int j = 0; j < 8; ++j) o[t][j] *= (j < 4) ? ca[j & 3] : cb[j & 3];
    // reload P in A-frag layout and accumulate O += P V
    AFrag pa;
    pa.h[0] = *(const bf16x8*)&pls[wv][r][16 * hi];
    pa.h[1] = *(const bf16x8*)&pls[wv][r][16 * hi + 8];
#pragma unroll
    for (int t = 0; t < 4; ++t) {                   // V^T is key-contiguous
      bf16x16 bv = *(const bf16x16*)(vptr + (size_t)(16 * t + r) * TT + c + 16 * hi);
      o[t] = wmma_bf16(pa.v, bv, o[t]);
    }
  };

  int c = 0;
  for (; c + 31 <= qbrow; c += 32) do_chunk(c, false);  // fully visible chunks
  for (; c < qbrow + 16; c += 32) do_chunk(c, true);    // exactly one edge chunk

  // broadcast 1/l back to C layout and store the O tile
  cls[wv][r] = 1.0f / rl;
  const f32x4 ia = ((const f32x4*)&cls[wv][8 * hi])[0];
  const f32x4 ib = ((const f32x4*)&cls[wv][8 * hi])[1];
#pragma unroll
  for (int j = 0; j < 8; ++j) {
    const float inv = (j < 4) ? ia[j & 3] : ib[j & 3];
    const int qrow = qbrow + j + 8 * hi;
#pragma unroll
    for (int t = 0; t < 4; ++t)
      attnb[((size_t)(b * TT + qrow)) * NE + h * HD + 16 * t + r] = (__bf16)(o[t][j] * inv);
  }
}

// ---------------- launcher ----------------
extern "C" void kernel_launch(void* const* d_in, const int* in_sizes, int n_in,
                              void* d_out, int out_size, void* d_ws, size_t ws_size,
                              hipStream_t stream) {
  const float* x   = (const float*)d_in[0];
  // d_in[1] = attn_mask (tril) -> causal mask computed analytically
  const float* Wq  = (const float*)d_in[2];
  const float* Wqb = (const float*)d_in[3];
  const float* Wk  = (const float*)d_in[4];
  const float* Wkb = (const float*)d_in[5];
  const float* Wv  = (const float*)d_in[6];
  const float* Wvb = (const float*)d_in[7];
  const float* Wo  = (const float*)d_in[8];
  const float* Wob = (const float*)d_in[9];
  float* out = (float*)d_out;

  char* p = (char*)d_ws;
  auto take = [&](size_t bytes) { char* q = p; p += (bytes + 255) & ~(size_t)255; return q; };
  __bf16* xb    = (__bf16*)take((size_t)MM * NE * 2);            // 12.6 MB (reused as attn)
  __bf16* wqb   = (__bf16*)take((size_t)NE * NE * 2);
  __bf16* wkb   = (__bf16*)take((size_t)(NKV * HD) * NE * 2);
  __bf16* wvb   = (__bf16*)take((size_t)(NKV * HD) * NE * 2);
  __bf16* wob   = (__bf16*)take((size_t)NE * NE * 2);
  __bf16* qbuf  = (__bf16*)take((size_t)BB * NH  * TT * HD * 2); // 12.6 MB
  __bf16* kbuf  = (__bf16*)take((size_t)BB * NKV * TT * HD * 2); //  6.3 MB
  __bf16* vtbuf = (__bf16*)take((size_t)BB * NKV * HD * TT * 2); //  6.3 MB
  __bf16* attn  = xb;  // x_bf16 is dead after the V projection; stream-ordered alias

  auto cvt = [&](const float* src, __bf16* dst, int n) {
    int n4 = n >> 2;
    cvt_f32_bf16<<<(n4 + 255) / 256, 256, 0, stream>>>(src, dst, n4);
  };
  cvt(x,  xb,  MM * NE);
  cvt(Wq, wqb, NE * NE);
  cvt(Wk, wkb, NKV * HD * NE);
  cvt(Wv, wvb, NKV * HD * NE);
  cvt(Wo, wob, NE * NE);

  // waves = (M/16)*(N/64); 8 waves per 256-thread block
  const int blkQ = ((MM / 16) * (NE / 64)) / 8;         // 768
  const int blkK = ((MM / 16) * ((NKV * HD) / 64)) / 8; // 384
  gemm_xWT<1><<<blkQ, 256, 0, stream>>>(xb, wqb, Wqb, qbuf,  MM, NE,       NE, NH);
  gemm_xWT<1><<<blkK, 256, 0, stream>>>(xb, wkb, Wkb, kbuf,  MM, NKV * HD, NE, NKV);
  gemm_xWT<2><<<blkK, 256, 0, stream>>>(xb, wvb, Wvb, vtbuf, MM, NKV * HD, NE, NKV);

  const int blkA = (BB * NH * (TT / 16)) / 8;           // 768
  flash_attn<<<blkA, 256, 0, stream>>>(qbuf, kbuf, vtbuf, attn,
      0.051031036307982884f * 1.4426950408889634f /* (1/sqrt(384)) * log2(e) */);

  gemm_xWT<0><<<blkQ, 256, 0, stream>>>(attn, wob, Wob, out, MM, NE, NE, NH);
}
